// Sequence_6227702579479
// MI455X (gfx1250) — matrix operand order repaired and checked by reading, compiled, not run on hardware
//
#include <hip/hip_runtime.h>

#define T_STEPS 512
#define BATCH   128
#define HID     512
#define DIN     5
#define NWG     32
#define HC      16   // hidden columns owned per workgroup (one 16-wide tile per gate)

// LDS layout: 3 matrices x 4 tiles x 16 k-chunks x 1KB fragment blocks, then c1/c2
#define WLDS_BYTES   (3 * 4 * 16 * 1024)          // 196608
#define C1_OFF       WLDS_BYTES
#define C2_OFF       (WLDS_BYTES + BATCH * HC * 4)
#define SMEM_BYTES   (WLDS_BYTES + 2 * BATCH * HC * 4)   // 212992

typedef __attribute__((ext_vector_type(16))) __bf16 v16bf;
typedef __attribute__((ext_vector_type(8)))  float  v8f;

union Frag { v16bf v; uint4 q[2]; };

__device__ __forceinline__ unsigned short f2bf(float f) {
    unsigned u = __float_as_uint(f);
    unsigned r = (u + 0x7FFFu + ((u >> 16) & 1u)) >> 16;   // RNE
    return (unsigned short)r;
}
__device__ __forceinline__ float bf2f(unsigned short h) {
    return __uint_as_float(((unsigned)h) << 16);
}
__device__ __forceinline__ float frcp(float x) { return __builtin_amdgcn_rcpf(x); }
__device__ __forceinline__ float sigm(float x) { return frcp(1.0f + __expf(-x)); }
__device__ __forceinline__ float tanh_f(float x) {
    x = fminf(fmaxf(x, -15.0f), 15.0f);           // keep e finite -> no inf/inf
    float e = __expf(2.0f * x);
    return (e - 1.0f) * frcp(e + 1.0f);
}

// Grid-wide sense barrier on L2 atomics. sync[0]=arrive count, sync[1]=generation.
__device__ __forceinline__ void grid_barrier(unsigned* sync) {
    __syncthreads();
    if (threadIdx.x == 0) {
        __threadfence();
        unsigned g = __hip_atomic_load(&sync[1], __ATOMIC_RELAXED, __HIP_MEMORY_SCOPE_AGENT);
        unsigned old = __hip_atomic_fetch_add(&sync[0], 1u, __ATOMIC_ACQ_REL, __HIP_MEMORY_SCOPE_AGENT);
        if (old == (unsigned)(NWG - 1)) {
            __hip_atomic_store(&sync[0], 0u, __ATOMIC_RELAXED, __HIP_MEMORY_SCOPE_AGENT);
            __hip_atomic_fetch_add(&sync[1], 1u, __ATOMIC_RELEASE, __HIP_MEMORY_SCOPE_AGENT);
        } else {
            while (__hip_atomic_load(&sync[1], __ATOMIC_ACQUIRE, __HIP_MEMORY_SCOPE_AGENT) == g) {
                __builtin_amdgcn_s_sleep(2);
            }
        }
        __threadfence();
    }
    __syncthreads();
}

__global__ void init_ws_kernel(unsigned short* hbufs, unsigned* sync, int n) {
    int i = blockIdx.x * blockDim.x + threadIdx.x;
    if (i < n) hbufs[i] = 0;
    if (i < 2) sync[i] = 0;
}

// Convert the three 2048x512 recurrent weight matrices fp32 -> bf16 (row-major).
__global__ void convert_weights(const float* __restrict__ a, const float* __restrict__ b,
                                const float* __restrict__ c, unsigned short* __restrict__ dst,
                                int n) {
    int i = blockIdx.x * blockDim.x + threadIdx.x;
    if (i >= 3 * n) return;
    float v;
    if (i < n)          v = a[i];
    else if (i < 2 * n) v = b[i - n];
    else                v = c[i - 2 * n];
    dst[i] = f2bf(v);
}

// K-loop over 16 chunks of 32. BOTH operands are software double-buffered:
// A (global) and B (LDS) each ping-pong between two register sets, so the 8
// ds_load_b128 of the next chunk are in flight while the current chunk's 4
// WMMAs execute. Two live B buffers prevent the scheduler from collapsing
// them into one register set (which would serialize load->wait->wmma).
#define GEMM_K_LOOP(APTR, WBASE)                                                   \
    {                                                                              \
        Frag a0, a1, bA[4], bB[4];                                                 \
        a0.q[0] = *(const uint4*)((APTR) + arow + hi * 8);                         \
        a0.q[1] = *(const uint4*)((APTR) + arow + 16 + hi * 8);                    \
        _Pragma("unroll")                                                          \
        for (int n = 0; n < 4; ++n) {                                              \
            int base = (WBASE) + (n * 16) * 1024;                                  \
            bA[n].q[0] = *(const uint4*)(smem + base + lane * 16);                 \
            bA[n].q[1] = *(const uint4*)(smem + base + 512 + lane * 16);           \
        }                                                                          \
        _Pragma("unroll 1")                                                        \
        for (int kci = 0; kci < 16; kci += 2) {                                    \
            {   int kc = (kci + 1) * 32;                                           \
                a1.q[0] = *(const uint4*)((APTR) + arow + kc + hi * 8);            \
                a1.q[1] = *(const uint4*)((APTR) + arow + kc + 16 + hi * 8);       \
            }                                                                      \
            _Pragma("unroll")                                                      \
            for (int n = 0; n < 4; ++n) {                                          \
                int base = (WBASE) + (n * 16 + kci + 1) * 1024;                    \
                bB[n].q[0] = *(const uint4*)(smem + base + lane * 16);             \
                bB[n].q[1] = *(const uint4*)(smem + base + 512 + lane * 16);       \
            }                                                                      \
            _Pragma("unroll")                                                      \
            for (int n = 0; n < 4; ++n)                                            \
                acc[n] = __builtin_amdgcn_wmma_f32_16x16x32_bf16(                  \
                    false, a0.v, false, bA[n].v, (short)0, acc[n], false, false);  \
            if (kci + 2 < 16) {                                                    \
                int kc = (kci + 2) * 32;                                           \
                a0.q[0] = *(const uint4*)((APTR) + arow + kc + hi * 8);            \
                a0.q[1] = *(const uint4*)((APTR) + arow + kc + 16 + hi * 8);       \
                _Pragma("unroll")                                                  \
                for (int n = 0; n < 4; ++n) {                                      \
                    int base = (WBASE) + (n * 16 + kci + 2) * 1024;                \
                    bA[n].q[0] = *(const uint4*)(smem + base + lane * 16);         \
                    bA[n].q[1] = *(const uint4*)(smem + base + 512 + lane * 16);   \
                }                                                                  \
            }                                                                      \
            _Pragma("unroll")                                                      \
            for (int n = 0; n < 4; ++n)                                            \
                acc[n] = __builtin_amdgcn_wmma_f32_16x16x32_bf16(                  \
                    false, a1.v, false, bB[n].v, (short)0, acc[n], false, false);  \
        }                                                                          \
    }

__global__ __launch_bounds__(256, 1)
void lstm_persistent(const float* __restrict__ x,
                     const float* __restrict__ Wih1,
                     const float* __restrict__ bih1, const float* __restrict__ bhh1,
                     const float* __restrict__ bih2, const float* __restrict__ bhh2,
                     const float* __restrict__ Wlin, const float* __restrict__ blin,
                     const unsigned short* __restrict__ whh1,
                     const unsigned short* __restrict__ wih2,
                     const unsigned short* __restrict__ whh2,
                     unsigned short* __restrict__ h1a, unsigned short* __restrict__ h1b,
                     unsigned short* __restrict__ h2a, unsigned short* __restrict__ h2b,
                     unsigned* __restrict__ sync,
                     float* __restrict__ out)
{
    extern __shared__ char smem[];          // 192KB swizzled weights + 16KB cell state
    float* c1s = (float*)(smem + C1_OFF);   // [BATCH][HC]
    float* c2s = (float*)(smem + C2_OFF);

    const int tid  = threadIdx.x;
    const int wg   = blockIdx.x;
    const int wave = tid >> 5;
    const int lane = tid & 31;
    const int lm   = lane & 15;         // N / M index within tile
    const int hi   = lane >> 4;         // which half of the wave
    const int mbase = wave * 16;        // 16 batch rows per wave

    // ---- one-time: stage this WG's weight slices into LDS, fragment-swizzled ----
    {
        const unsigned short* mats[3] = { whh1, wih2, whh2 };
        for (int blk = wave; blk < 3 * 4 * 16; blk += 8) {
            int w = blk >> 6;
            int n = (blk >> 4) & 3;
            int c = blk & 15;
            const unsigned short* src =
                mats[w] + (size_t)(n * HID + wg * HC + lm) * HID + c * 32 + hi * 8;
            uint4 q0 = *(const uint4*)src;
            uint4 q1 = *(const uint4*)(src + 16);
            *(uint4*)(smem + (size_t)blk * 1024 + lane * 16)       = q0;
            *(uint4*)(smem + (size_t)blk * 1024 + 512 + lane * 16) = q1;
        }
    }
    for (int i = tid; i < BATCH * HC; i += 256) { c1s[i] = 0.0f; c2s[i] = 0.0f; }
    __syncthreads();

    const float blin0 = blin[0];
    const int arow = (mbase + lm) * HID;   // A-fragment row base (bf16 elements)
    const int colbase = wg * HC + lm;      // this lane's gate column (per 512-block)

#pragma unroll 1
    for (int t = 0; t < T_STEPS; ++t) {
        const unsigned short* h1r = (t & 1) ? h1b : h1a;
        unsigned short*       h1w = (t & 1) ? h1a : h1b;
        const unsigned short* h2r = (t & 1) ? h2b : h2a;
        unsigned short*       h2w = (t & 1) ? h2a : h2b;

        // ================= Layer 1 =================
        v8f acc[4];
        {   // init accumulators with bias + x_t @ Wih1^T (K=5, scalar VALU)
            float xv[8][DIN];
#pragma unroll
            for (int r = 0; r < 8; ++r) {
                int m = mbase + r + 8 * hi;
                const float* xp = x + ((size_t)t * BATCH + m) * DIN;
#pragma unroll
                for (int k = 0; k < DIN; ++k) xv[r][k] = xp[k];
            }
#pragma unroll
            for (int n = 0; n < 4; ++n) {
                int col = n * HID + colbase;
                const float* wp = Wih1 + col * DIN;
                float w0 = wp[0], w1 = wp[1], w2 = wp[2], w3 = wp[3], w4 = wp[4];
                float bias = bih1[col] + bhh1[col];
#pragma unroll
                for (int r = 0; r < 8; ++r) {
                    acc[n][r] = bias + xv[r][0]*w0 + xv[r][1]*w1 + xv[r][2]*w2
                                     + xv[r][3]*w3 + xv[r][4]*w4;
                }
            }
        }
        GEMM_K_LOOP(h1r, 0)                 // + h1 @ Whh1^T
        // LSTM cell nonlinearity -> h1w, c1s
#pragma unroll
        for (int r = 0; r < 8; ++r) {
            int m = mbase + r + 8 * hi;
            float cn = sigm(acc[1][r]) * c1s[m * HC + lm]
                     + sigm(acc[0][r]) * tanh_f(acc[2][r]);
            c1s[m * HC + lm] = cn;
            h1w[m * HID + wg * HC + lm] = f2bf(sigm(acc[3][r]) * tanh_f(cn));
        }
        grid_barrier(sync);   // publish h1(t)

        // ================= Layer 2 =================
#pragma unroll
        for (int n = 0; n < 4; ++n) {
            int col = n * HID + colbase;
            float bias = bih2[col] + bhh2[col];
#pragma unroll
            for (int r = 0; r < 8; ++r) acc[n][r] = bias;
        }
        GEMM_K_LOOP(h1w, 1 * 4 * 16 * 1024) // + h1_new @ Wih2^T
        GEMM_K_LOOP(h2r, 2 * 4 * 16 * 1024) // + h2_old @ Whh2^T
#pragma unroll
        for (int r = 0; r < 8; ++r) {
            int m = mbase + r + 8 * hi;
            float cn = sigm(acc[1][r]) * c2s[m * HC + lm]
                     + sigm(acc[0][r]) * tanh_f(acc[2][r]);
            c2s[m * HC + lm] = cn;
            h2w[m * HID + wg * HC + lm] = f2bf(sigm(acc[3][r]) * tanh_f(cn));
        }
        grid_barrier(sync);   // publish h2(t)

        // ============ Output head: WGs 0..15, wave handles batch b = wg*8 + wave ============
        if (wg < 16) {
            int b = wg * 8 + wave;
            float part = 0.0f;
#pragma unroll
            for (int it = 0; it < 16; ++it) {
                int j = lane + it * 32;
                part += bf2f(h2w[b * HID + j]) * Wlin[j];
            }
#pragma unroll
            for (int off = 16; off > 0; off >>= 1) part += __shfl_xor(part, off, 32);
            if (lane == 0) out[(size_t)b * T_STEPS + t] = part + blin0;
        }
    }
}

extern "C" void kernel_launch(void* const* d_in, const int* in_sizes, int n_in,
                              void* d_out, int out_size, void* d_ws, size_t ws_size,
                              hipStream_t stream) {
    const float* x    = (const float*)d_in[0];
    const float* Wih1 = (const float*)d_in[1];
    const float* Whh1 = (const float*)d_in[2];
    const float* bih1 = (const float*)d_in[3];
    const float* bhh1 = (const float*)d_in[4];
    const float* Wih2 = (const float*)d_in[5];
    const float* Whh2 = (const float*)d_in[6];
    const float* bih2 = (const float*)d_in[7];
    const float* bhh2 = (const float*)d_in[8];
    const float* Wlin = (const float*)d_in[9];
    const float* blin = (const float*)d_in[10];

    char* ws = (char*)d_ws;
    unsigned* sync = (unsigned*)ws;                          // 8 bytes
    unsigned short* hbase = (unsigned short*)(ws + 4096);    // 4 x 128x512 bf16 buffers
    unsigned short* h1a = hbase;
    unsigned short* h1b = hbase + 65536;
    unsigned short* h2a = hbase + 2 * 65536;
    unsigned short* h2b = hbase + 3 * 65536;
    unsigned short* whh1b = (unsigned short*)(ws + (1u << 20)); // 3 x 2MB bf16 weights
    unsigned short* wih2b = whh1b + 1048576;
    unsigned short* whh2b = whh1b + 2 * 1048576;

    init_ws_kernel<<<1024, 256, 0, stream>>>(hbase, sync, 4 * 65536);
    convert_weights<<<(3 * 1048576 + 255) / 256, 256, 0, stream>>>(Whh1, Wih2, Whh2, whh1b, 1048576);
    lstm_persistent<<<NWG, 256, SMEM_BYTES, stream>>>(x, Wih1, bih1, bhh1, bih2, bhh2, Wlin, blin,
                                                      whh1b, wih2b, whh2b,
                                                      h1a, h1b, h2a, h2b, sync, (float*)d_out);
}
